// Autoregression_autoregression_54374285967487
// MI455X (gfx1250) — matrix operand order repaired and checked by reading, not compiled, not run on hardware
//
#include <hip/hip_runtime.h>

#define NJ 23

typedef __attribute__((ext_vector_type(16))) __bf16 v16bf;
typedef __attribute__((ext_vector_type(8)))  __bf16 v8bf;
typedef __attribute__((ext_vector_type(8)))  float  v8f;

// ---- ancestor topology (from reference _ancestors()) ----
__constant__ int c_anc_off[NJ + 1] = {
    0, 0, 0, 0, 1, 2, 3, 5, 7, 9, 12, 15, 18, 21, 24, 28, 32, 36, 41, 46, 52, 58, 65, 72};
__constant__ int c_anc_list[72] = {
    0, 1, 2,
    3, 0,  4, 1,  5, 2,
    6, 3, 0,  7, 4, 1,  8, 5, 2,
    8, 5, 2,  8, 5, 2,
    11, 8, 5, 2,  12, 8, 5, 2,  13, 8, 5, 2,
    15, 12, 8, 5, 2,  16, 13, 8, 5, 2,
    17, 15, 12, 8, 5, 2,  18, 16, 13, 8, 5, 2,
    19, 17, 15, 12, 8, 5, 2,  20, 18, 16, 13, 8, 5, 2};
// K-tiles (of 32) per joint layer-1:  din = 64 + 21*na
__constant__ int c_nkt[NJ] = {2,2,2, 3,3,3, 4,4,4, 4,4,4,4,4, 5,5,5, 6,6, 6,6, 7,7};
// base tile of W1[j] in packed workspace (tile = 512 bf16); W_embed uses tiles 0..11
__constant__ int c_w1_base[NJ] = {12,16,20, 24,30,36, 42,50,58, 66,74,82,90,98,
                                  106,116,126, 136,148, 160,172, 184,198};
#define W2_BASE_TILE 212          // W2[j] at tile 212+j
#define N_TILES      235
#define GATHER_OFF   (N_TILES * 512)   // bf16-element offset of gather table in ws
// gather-table base per joint ((nkt-2)*32 entries each)
__constant__ int c_gath_base[NJ + 1] = {0,0,0, 0,32,64, 96,160,224, 288,352,416,480,544,
                                        608,704,800, 896,1024, 1152,1280, 1408,1568, 1728};
#define HIST_W  484               // 23*21 features + 1 zero pad slot
#define ZSLOT   483

struct KParams {
    const float* glob;
    const float* W_embed;
    const float* b_embed;
    const float* W1[NJ];
    const float* b1[NJ];
    const float* W2[NJ];
    const float* b2[NJ];
    float*       out;
    __bf16*      ws;
    int          Brows;
};

__device__ __forceinline__ __bf16 f2bf(float f) {
    union { float f; unsigned u; } x; x.f = f;
    unsigned r = x.u + 0x7FFFu + ((x.u >> 16) & 1u);   // round-to-nearest-even
    unsigned short h = (unsigned short)(r >> 16);
    __bf16 o; __builtin_memcpy(&o, &h, 2); return o;
}
__device__ __forceinline__ float elu(float x) { return x > 0.f ? x : (__expf(x) - 1.f); }
__device__ __forceinline__ float det3(const float m[9]) {
    return m[0] * (m[4] * m[8] - m[5] * m[7])
         - m[1] * (m[3] * m[8] - m[5] * m[6])
         + m[2] * (m[3] * m[7] - m[4] * m[6]);
}
__device__ __forceinline__ void jacobi(float A[9], float V[9], int p, int q) {
    float apq = A[3 * p + q];
    if (__builtin_fabsf(apq) > 1e-12f) {
        float app = A[3 * p + p], aqq = A[3 * q + q];
        float tau = (aqq - app) / (2.f * apq);
        float t = __builtin_copysignf(1.f, tau) / (__builtin_fabsf(tau) + __builtin_sqrtf(1.f + tau * tau));
        float c = __frsqrt_rn(1.f + t * t);
        float s = t * c;
        #pragma unroll
        for (int k = 0; k < 3; k++) {
            float akp = A[3 * k + p], akq = A[3 * k + q];
            A[3 * k + p] = c * akp - s * akq;
            A[3 * k + q] = s * akp + c * akq;
        }
        #pragma unroll
        for (int k = 0; k < 3; k++) {
            float apk = A[3 * p + k], aqk = A[3 * q + k];
            A[3 * p + k] = c * apk - s * aqk;
            A[3 * q + k] = s * apk + c * aqk;
        }
        #pragma unroll
        for (int k = 0; k < 3; k++) {
            float vkp = V[3 * k + p], vkq = V[3 * k + q];
            V[3 * k + p] = c * vkp - s * vkq;
            V[3 * k + q] = s * vkp + c * vkq;
        }
    }
}
__device__ __forceinline__ v8f wmma_bf16(v16bf a, v16bf b, v8f c) {
    return __builtin_amdgcn_wmma_f32_16x16x32_bf16(false, a, false, b, (short)0, c, false, false);
}
// one aligned 32B fragment load from the packed weight workspace
__device__ __forceinline__ v16bf load_tile(const __bf16* ws, int tile, int lane) {
    return *(const v16bf*)(ws + ((long)tile * 32 + lane) * 16);
}
// A-fragment from an LDS row (ISA 7.12.2, 16-bit A 16x32): two aligned 16B chunks
__device__ __forceinline__ v16bf a_frag_lds(const __bf16* rowBase, int k0, int khalfA) {
    v8bf lo = *(const v8bf*)(rowBase + k0 + khalfA);
    v8bf hi = *(const v8bf*)(rowBase + k0 + 16 + khalfA);
    return __builtin_shufflevector(lo, hi, 0,1,2,3,4,5,6,7,8,9,10,11,12,13,14,15);
}

// ---------------- prep: pack weights (bf16, fragment order) + gather table into ws -----------
__global__ __launch_bounds__(256)
void prep_pack_kernel(KParams P) {
    __bf16* ws = P.ws;
    int t = blockIdx.x, tid = threadIdx.x;
    if (t < N_TILES) {
        const float* src; int kt, nt, Kv, stride, Nv;
        if (t < 12)               { src = P.W_embed; kt = t >> 2; nt = t & 3; Kv = 72; stride = 64; Nv = 64; }
        else if (t < W2_BASE_TILE) {
            int j = 0;
            while (j + 1 < NJ && c_w1_base[j + 1] <= t) j++;
            int local = t - c_w1_base[j];
            kt = local >> 1; nt = local & 1;
            src = P.W1[j]; Kv = 64 + 21 * (c_anc_off[j + 1] - c_anc_off[j]); stride = 32; Nv = 32;
        } else                    { src = P.W2[t - W2_BASE_TILE]; kt = 0; nt = 0; Kv = 32; stride = 9; Nv = 9; }
        for (int i = tid; i < 512; i += 256) {
            int l = i >> 4, e = i & 15;
            int k = kt * 32 + ((e < 8) ? 0 : 16) + ((l < 16) ? 0 : 8) + (e & 7);
            int n = nt * 16 + (l & 15);
            float v = (k < Kv && n < Nv) ? src[k * stride + n] : 0.f;
            ws[(long)t * 512 + i] = f2bf(v);
        }
    } else {  // ancestor-gather index table
        unsigned short* gt = (unsigned short*)(ws + GATHER_OFF);
        for (int i = tid; i < 1728; i += 256) {
            int j = 0;
            while (j + 1 < NJ && c_gath_base[j + 1] <= i) j++;
            int kf = i - c_gath_base[j];
            int na = c_anc_off[j + 1] - c_anc_off[j];
            const int* par = &c_anc_list[c_anc_off[j]];
            unsigned short off = ZSLOT;
            if (kf < 21 * na) {
                int p, f;
                if (kf < 9 * na)       { p = par[kf / 9]; f = kf % 9; }
                else if (kf < 12 * na) { int q = kf - 9 * na;  p = par[q / 3]; f = 9 + q % 3; }
                else                   { int q = kf - 12 * na; p = par[q / 9]; f = 12 + q % 9; }
                off = (unsigned short)(p * 21 + f);
            }
            gt[i] = off;
        }
    }
}

// ---------------- main: one wave32 per block owns 32 batch rows for the full chain -----------
__global__ __launch_bounds__(32)
void Autoregression_autoregression_54374285967487_kernel(KParams P) {
    __shared__ __attribute__((aligned(16))) __bf16 s_hist[32][HIST_W];  // U_p|S_p|rot per joint
    __shared__ __attribute__((aligned(16))) __bf16 s_stage[32 * 96];    // glob / embed / h / F staging
    __shared__ unsigned short s_off[160];

    const __bf16* ws = P.ws;
    const int lane   = threadIdx.x;
    const int b0     = blockIdx.x * 32;
    const int n_     = lane & 15;
    const int khalfA = (lane < 16) ? 0 : 8;
    const int mhalf  = (lane < 16) ? 0 : 8;

    // ---- stage glob rows (coalesced float4) into LDS as bf16, zero-padded to K=96 ----
    {
        const float4* g4 = (const float4*)(P.glob + (long)(b0 + lane) * 72);  // 288B rows, 16B aligned
        __bf16* row = &s_stage[lane * 96];
        #pragma unroll
        for (int i = 0; i < 18; i++) {
            float4 w = g4[i];
            row[4 * i + 0] = f2bf(w.x); row[4 * i + 1] = f2bf(w.y);
            row[4 * i + 2] = f2bf(w.z); row[4 * i + 3] = f2bf(w.w);
        }
        #pragma unroll
        for (int i = 72; i < 96; i++) row[i] = f2bf(0.f);
        s_hist[lane][ZSLOT] = f2bf(0.f);     // zero slot for OOB gather
    }
    __syncthreads();

    // ---- embed = elu(glob @ W_embed + b): M=32, N=64, K=96(padded) ----
    v8f ce[2][4];
    #pragma unroll
    for (int mt = 0; mt < 2; mt++)
        #pragma unroll
        for (int nt = 0; nt < 4; nt++) {
            float bb = P.b_embed[nt * 16 + n_];
            #pragma unroll
            for (int v = 0; v < 8; v++) ce[mt][nt][v] = bb;
        }
    #pragma unroll
    for (int kt = 0; kt < 3; kt++) {
        v16bf afr[2];
        #pragma unroll
        for (int mt = 0; mt < 2; mt++)
            afr[mt] = a_frag_lds(&s_stage[(mt * 16 + n_) * 96], kt * 32, khalfA);
        #pragma unroll
        for (int nt = 0; nt < 4; nt++) {
            v16bf bt = load_tile(ws, kt * 4 + nt, lane);
            ce[0][nt] = wmma_bf16(afr[0], bt, ce[0][nt]);
            ce[1][nt] = wmma_bf16(afr[1], bt, ce[1][nt]);
        }
    }
    __syncthreads();
    #pragma unroll
    for (int mt = 0; mt < 2; mt++)
        #pragma unroll
        for (int nt = 0; nt < 4; nt++)
            #pragma unroll
            for (int v = 0; v < 8; v++) {
                int m = mt * 16 + mhalf + v;
                s_stage[m * 64 + nt * 16 + n_] = f2bf(elu(ce[mt][nt][v]));
            }
    __syncthreads();
    v16bf embFrag[2][2];                       // persistent embed A-fragments, reused by all joints
    #pragma unroll
    for (int mt = 0; mt < 2; mt++)
        #pragma unroll
        for (int kt = 0; kt < 2; kt++)
            embFrag[mt][kt] = a_frag_lds(&s_stage[(mt * 16 + n_) * 64], kt * 32, khalfA);
    __syncthreads();

    // ---------------- 23-joint autoregressive chain ----------------
    const unsigned short* gt = (const unsigned short*)(ws + GATHER_OFF);
    for (int j = 0; j < NJ; j++) {
        const int nkt  = c_nkt[j];
        const int gcnt = (nkt - 2) * 32;
        const int w1b  = c_w1_base[j];
        for (int i = lane; i < gcnt; i += 32) s_off[i] = gt[c_gath_base[j] + i];
        if (j + 1 < NJ) __builtin_prefetch(ws + (long)c_w1_base[j + 1] * 512, 0, 1);
        __syncthreads();   // also orders previous joint's hist writes vs this joint's gathers

        // ---- layer1: (32 x din) @ (din x 32) ----
        v8f c1[2][2];
        #pragma unroll
        for (int mt = 0; mt < 2; mt++)
            #pragma unroll
            for (int nt = 0; nt < 2; nt++) {
                float bb = P.b1[j][nt * 16 + n_];
                #pragma unroll
                for (int v = 0; v < 8; v++) c1[mt][nt][v] = bb;
            }
        #pragma unroll
        for (int kt = 0; kt < 2; kt++) {       // embed K-tiles from registers
            v16bf bt0 = load_tile(ws, w1b + kt * 2 + 0, lane);
            v16bf bt1 = load_tile(ws, w1b + kt * 2 + 1, lane);
            c1[0][0] = wmma_bf16(embFrag[0][kt], bt0, c1[0][0]);
            c1[1][0] = wmma_bf16(embFrag[1][kt], bt0, c1[1][0]);
            c1[0][1] = wmma_bf16(embFrag[0][kt], bt1, c1[0][1]);
            c1[1][1] = wmma_bf16(embFrag[1][kt], bt1, c1[1][1]);
        }
        for (int kt = 2; kt < nkt; kt++) {     // ancestor K-tiles gathered from LDS history
            unsigned short idxs[16];
            #pragma unroll
            for (int e = 0; e < 16; e++)
                idxs[e] = s_off[(kt - 2) * 32 + ((e < 8) ? 0 : 16) + khalfA + (e & 7)];
            v16bf afr[2];
            #pragma unroll
            for (int mt = 0; mt < 2; mt++) {
                const __bf16* hrow = &s_hist[mt * 16 + n_][0];
                #pragma unroll
                for (int e = 0; e < 16; e++) afr[mt][e] = hrow[idxs[e]];
            }
            v16bf bt0 = load_tile(ws, w1b + kt * 2 + 0, lane);
            v16bf bt1 = load_tile(ws, w1b + kt * 2 + 1, lane);
            c1[0][0] = wmma_bf16(afr[0], bt0, c1[0][0]);
            c1[1][0] = wmma_bf16(afr[1], bt0, c1[1][0]);
            c1[0][1] = wmma_bf16(afr[0], bt1, c1[0][1]);
            c1[1][1] = wmma_bf16(afr[1], bt1, c1[1][1]);
        }
        // elu -> bf16 staging for layer-2 A transpose
        #pragma unroll
        for (int mt = 0; mt < 2; mt++)
            #pragma unroll
            for (int nt = 0; nt < 2; nt++)
                #pragma unroll
                for (int v = 0; v < 8; v++) {
                    int m = mt * 16 + mhalf + v;
                    s_stage[m * 32 + nt * 16 + n_] = f2bf(elu(c1[mt][nt][v]));
                }
        __syncthreads();

        // ---- layer2: (32 x 32) @ (32 x 9 padded to 16) ----
        v8f c2[2];
        {
            float bb = (n_ < 9) ? P.b2[j][n_] : 0.f;
            #pragma unroll
            for (int mt = 0; mt < 2; mt++)
                #pragma unroll
                for (int v = 0; v < 8; v++) c2[mt][v] = bb;
            v16bf bt = load_tile(ws, W2_BASE_TILE + j, lane);
            #pragma unroll
            for (int mt = 0; mt < 2; mt++) {
                v16bf afr = a_frag_lds(&s_stage[(mt * 16 + n_) * 32], 0, khalfA);
                c2[mt] = wmma_bf16(afr, bt, c2[mt]);
            }
        }
        __syncthreads();
        float* fS = (float*)(s_stage + 1024);  // [32][9] f32, disjoint from h region
        #pragma unroll
        for (int mt = 0; mt < 2; mt++)
            #pragma unroll
            for (int v = 0; v < 8; v++) {
                int m = mt * 16 + mhalf + v;
                if (n_ < 9) fS[m * 9 + n_] = c2[mt][v];
            }
        __syncthreads();

        // ---- one 3x3 SVD per lane (row = lane) ----
        float F[9];
        #pragma unroll
        for (int i = 0; i < 9; i++) F[i] = fS[lane * 9 + i];

        float A[9], V[9] = {1, 0, 0, 0, 1, 0, 0, 0, 1};
        #pragma unroll
        for (int r = 0; r < 3; r++)
            #pragma unroll
            for (int c = 0; c < 3; c++) {
                float s = 0.f;
                #pragma unroll
                for (int k = 0; k < 3; k++) s += F[3 * k + r] * F[3 * k + c];
                A[3 * r + c] = s;
            }
        for (int sweep = 0; sweep < 5; sweep++) {
            jacobi(A, V, 0, 1); jacobi(A, V, 0, 2); jacobi(A, V, 1, 2);
        }
        float lam[3] = {A[0], A[4], A[8]};
        #pragma unroll
        for (int a_ = 0; a_ < 2; a_++)
            #pragma unroll
            for (int i = 0; i < 2 - a_; i++)
                if (lam[i] < lam[i + 1]) {
                    float t = lam[i]; lam[i] = lam[i + 1]; lam[i + 1] = t;
                    #pragma unroll
                    for (int r = 0; r < 3; r++) {
                        float tv = V[3 * r + i]; V[3 * r + i] = V[3 * r + i + 1]; V[3 * r + i + 1] = tv;
                    }
                }
        float S3[3];
        #pragma unroll
        for (int i = 0; i < 3; i++) S3[i] = __builtin_sqrtf(__builtin_fmaxf(lam[i], 0.f));

        float u[3][3];
        #pragma unroll
        for (int c = 0; c < 3; c++)
            #pragma unroll
            for (int r = 0; r < 3; r++)
                u[c][r] = F[3 * r + 0] * V[0 + c] + F[3 * r + 1] * V[3 + c] + F[3 * r + 2] * V[6 + c];
        float nn = __builtin_sqrtf(u[0][0] * u[0][0] + u[0][1] * u[0][1] + u[0][2] * u[0][2]);
        if (nn > 1e-8f) { float inv = 1.f / nn; u[0][0] *= inv; u[0][1] *= inv; u[0][2] *= inv; }
        else            { u[0][0] = 1.f; u[0][1] = 0.f; u[0][2] = 0.f; }
        float d01 = u[1][0] * u[0][0] + u[1][1] * u[0][1] + u[1][2] * u[0][2];
        u[1][0] -= d01 * u[0][0]; u[1][1] -= d01 * u[0][1]; u[1][2] -= d01 * u[0][2];
        nn = __builtin_sqrtf(u[1][0] * u[1][0] + u[1][1] * u[1][1] + u[1][2] * u[1][2]);
        if (nn > 1e-8f) { float inv = 1.f / nn; u[1][0] *= inv; u[1][1] *= inv; u[1][2] *= inv; }
        else {
            float ex = (__builtin_fabsf(u[0][0]) < 0.9f) ? 1.f : 0.f;
            float ey = 1.f - ex;
            u[1][0] = -u[0][2] * ey;
            u[1][1] =  u[0][2] * ex;
            u[1][2] =  u[0][0] * ey - u[0][1] * ex;
            float inv = __frsqrt_rn(u[1][0] * u[1][0] + u[1][1] * u[1][1] + u[1][2] * u[1][2] + 1e-30f);
            u[1][0] *= inv; u[1][1] *= inv; u[1][2] *= inv;
        }
        u[2][0] = u[0][1] * u[1][2] - u[0][2] * u[1][1];
        u[2][1] = u[0][2] * u[1][0] - u[0][0] * u[1][2];
        u[2][2] = u[0][0] * u[1][1] - u[0][1] * u[1][0];

        float Umat[9];
        #pragma unroll
        for (int r = 0; r < 3; r++)
            #pragma unroll
            for (int c = 0; c < 3; c++) Umat[3 * r + c] = u[c][r];

        float dU = det3(Umat) >= 0.f ? 1.f : -1.f;
        float dV = det3(V)    >= 0.f ? 1.f : -1.f;
        float Up[9], Vp[9];
        #pragma unroll
        for (int r = 0; r < 3; r++) {
            Up[3 * r + 0] = Umat[3 * r + 0]; Up[3 * r + 1] = Umat[3 * r + 1]; Up[3 * r + 2] = Umat[3 * r + 2] * dU;
            Vp[3 * r + 0] = V[3 * r + 0];    Vp[3 * r + 1] = V[3 * r + 1];    Vp[3 * r + 2] = V[3 * r + 2] * dV;
        }
        float Sp2 = S3[2] * dU * dV;
        float R[9];
        #pragma unroll
        for (int r = 0; r < 3; r++)
            #pragma unroll
            for (int c = 0; c < 3; c++)
                R[3 * r + c] = Up[3 * r + 0] * Vp[3 * c + 0] + Up[3 * r + 1] * Vp[3 * c + 1] + Up[3 * r + 2] * Vp[3 * c + 2];

        // publish features for descendants (history in LDS)
        __bf16* hw = &s_hist[lane][j * 21];
        #pragma unroll
        for (int i = 0; i < 9; i++) hw[i] = f2bf(Up[i]);
        hw[9]  = f2bf(S3[0]);
        hw[10] = f2bf(S3[1]);
        hw[11] = f2bf(Sp2);
        #pragma unroll
        for (int i = 0; i < 9; i++) hw[12 + i] = f2bf(R[i]);

        // ---- streaming (non-temporal) outputs: F, U, S, V ----
        long b     = b0 + lane;
        long base9 = ((long)b * NJ + j) * 9;
        long base3 = ((long)b * NJ + j) * 3;
        long NB9   = (long)P.Brows * NJ * 9;
        long NB3   = (long)P.Brows * NJ * 3;
        float* out = P.out;
        #pragma unroll
        for (int i = 0; i < 9; i++) __builtin_nontemporal_store(F[i],    out + base9 + i);
        #pragma unroll
        for (int i = 0; i < 9; i++) __builtin_nontemporal_store(Umat[i], out + NB9 + base9 + i);
        #pragma unroll
        for (int i = 0; i < 3; i++) __builtin_nontemporal_store(S3[i],   out + 2 * NB9 + base3 + i);
        #pragma unroll
        for (int i = 0; i < 9; i++) __builtin_nontemporal_store(V[i],    out + 2 * NB9 + NB3 + base9 + i);
        __syncthreads();
    }
}

extern "C" void kernel_launch(void* const* d_in, const int* in_sizes, int n_in,
                              void* d_out, int out_size, void* d_ws, size_t ws_size,
                              hipStream_t stream) {
    KParams P;
    P.glob    = (const float*)d_in[0];
    P.W_embed = (const float*)d_in[1];
    P.b_embed = (const float*)d_in[2];
    for (int j = 0; j < NJ; j++) {
        P.W1[j] = (const float*)d_in[3 + j];
        P.b1[j] = (const float*)d_in[3 + NJ + j];
        P.W2[j] = (const float*)d_in[3 + 2 * NJ + j];
        P.b2[j] = (const float*)d_in[3 + 3 * NJ + j];
    }
    P.out   = (float*)d_out;
    P.ws    = (__bf16*)d_ws;
    P.Brows = in_sizes[0] / 72;

    // pack weights + gather table (235 weight tiles + 1 table block)
    hipLaunchKernelGGL(prep_pack_kernel, dim3(N_TILES + 1), dim3(256), 0, stream, P);
    // main chain: 32 batch rows per wave32 block
    hipLaunchKernelGGL(Autoregression_autoregression_54374285967487_kernel,
                       dim3(P.Brows / 32), dim3(32), 0, stream, P);
}
